// GraphADT_6768868459336
// MI455X (gfx1250) — matrix operand order repaired and checked by reading, not compiled, not run on hardware
//
#include <hip/hip_runtime.h>
#include <math.h>

// ---------------------------------------------------------------------------
// MI455X (gfx1250) implementation. All linear layers use
// v_wmma_f32_16x16x32_bf16 (wave32 WMMA, f32 accumulate), with B128
// vectorized fragment loads (fragment runs are 16B-aligned by construction).
// ---------------------------------------------------------------------------

typedef __attribute__((ext_vector_type(16))) __bf16 v16bf;
typedef __attribute__((ext_vector_type(8)))  __bf16 v8bf;
typedef __attribute__((ext_vector_type(8)))  float  v8f;

#define NB    128      // graphs per batch
#define NN0   512      // nodes per graph, level 0
#define NEDGE 524288   // edges per branch
#define DF    128      // feature width after convs
#define KP1   410      // ceil(0.8*512)
#define KP2   328      // ceil(0.8*410)
#define KP3   263      // ceil(0.8*328)

#define LB 256
static inline unsigned nblk(long long n) { return (unsigned)((n + LB - 1) / LB); }

// ------------------------------- utility kernels ---------------------------

__global__ void k_zero_f32(float* p, long long n) {
  long long i = (long long)blockIdx.x * blockDim.x + threadIdx.x;
  if (i < n) p[i] = 0.0f;
}

__global__ void k_copy_edges(const int* __restrict__ sin, const int* __restrict__ din,
                             int* __restrict__ s, int* __restrict__ d, int* __restrict__ v, int n) {
  int i = blockIdx.x * blockDim.x + threadIdx.x;
  if (i < n) { s[i] = sin[i]; d[i] = din[i]; v[i] = 1; }
}

// agg[dst] += x[src]  (valid edges only)
__global__ void k_scatter_add(const float* __restrict__ x, int ldx,
                              const int* __restrict__ src, const int* __restrict__ dst,
                              const int* __restrict__ valid,
                              float* __restrict__ agg, int lda, int nE, int feat) {
  long long tid = (long long)blockIdx.x * blockDim.x + threadIdx.x;
  long long tot = (long long)nE * feat;
  if (tid >= tot) return;
  int e = (int)(tid / feat);
  int f = (int)(tid - (long long)e * feat);
  if (!valid[e]) return;
  float vv = x[(long long)src[e] * ldx + f];
  atomicAdd(&agg[(long long)dst[e] * lda + f], vv);
}

// hb = bf16(x (+ agg)), zero padded to ldh columns
__global__ void k_build_hb(const float* __restrict__ x, int ldx,
                           const float* __restrict__ agg, int lda,
                           __bf16* __restrict__ hb, int ldh, long long rows, int feat) {
  long long tid = (long long)blockIdx.x * blockDim.x + threadIdx.x;
  long long tot = rows * ldh;
  if (tid >= tot) return;
  long long r = tid / ldh;
  int f = (int)(tid - r * ldh);
  float v = 0.0f;
  if (f < feat) {
    v = x[r * ldx + f];
    if (agg) v += agg[r * lda + f];
  }
  hb[tid] = (__bf16)v;
}

// weight [nout, kin] f32 row-major -> [npad, kpad] bf16, zero padded
__global__ void k_conv_w(const float* __restrict__ w, __bf16* __restrict__ wb,
                         int nout, int kin, int kpad, int npad) {
  long long tid = (long long)blockIdx.x * blockDim.x + threadIdx.x;
  long long tot = (long long)npad * kpad;
  if (tid >= tot) return;
  int nrow = (int)(tid / kpad);
  int k = (int)(tid - (long long)nrow * kpad);
  float v = (nrow < nout && k < kin) ? w[(long long)nrow * kin + k] : 0.0f;
  wb[tid] = (__bf16)v;
}

// ------------------------------- WMMA GEMM ---------------------------------
// C[M,Nvalid] = relu?( A[M,Kpad](bf16) * Bw[Npad,Kpad]^T(bf16) + bias )
// One wave per 16x16 output tile; 8 waves per block. A rows are clamped
// instead of guarded: out-of-range accumulator rows are simply never stored,
// so the K-loop has no divergent control flow (EXEC all-ones for WMMA) and
// all fragment loads are contiguous 16B-aligned b128 loads.
__global__ void k_gemm(const __bf16* __restrict__ A, int lda,
                       const __bf16* __restrict__ Bw, int ldb,
                       const float* __restrict__ bias,
                       float* __restrict__ C, int ldc,
                       int M, int Npad, int Nvalid, int Kpad, int relu) {
  int lane = threadIdx.x;                 // 0..31
  int wave = threadIdx.y;                 // 0..7
  int ntN = Npad >> 4;
  int ntM = (M + 15) >> 4;
  long long tile = (long long)blockIdx.x * 8 + wave;
  if (tile >= (long long)ntM * ntN) return;   // whole wave exits -> EXEC all-ones
  int tm = (int)(tile / ntN);
  int tn = (int)(tile - (long long)tm * ntN);

  int ml   = lane & 15;
  int arow = tm * 16 + ml;
  if (arow >= M) arow = M - 1;            // clamp: rows >= M never stored
  int aoff = (lane >= 16) ? 8 : 0;        // A fragment K base per lane group
  int boff = (lane >= 16) ? 8 : 0;        // B fragment K base per lane group
  const __bf16* Ar = A + (long long)arow * lda + aoff;
  const __bf16* Br = Bw + (long long)(tn * 16 + ml) * ldb + boff;

  v8f acc;
#pragma unroll
  for (int r = 0; r < 8; r++) acc[r] = 0.0f;

  for (int k0 = 0; k0 < Kpad; k0 += 32) {
    // A 16x32 bf16 layout: per lane two contiguous 8-elem runs (k0, k0+16)
    v8bf a0 = *(const v8bf*)(Ar + k0);
    v8bf a1 = *(const v8bf*)(Ar + k0 + 16);
    // B 32x16 bf16 layout: per lane two contiguous 8-elem runs (k0, k0+16)
    v8bf b0 = *(const v8bf*)(Br + k0);
    v8bf b1 = *(const v8bf*)(Br + k0 + 16);
    v16bf a = __builtin_shufflevector(a0, a1, 0, 1, 2, 3, 4, 5, 6, 7,
                                      8, 9, 10, 11, 12, 13, 14, 15);
    v16bf b = __builtin_shufflevector(b0, b1, 0, 1, 2, 3, 4, 5, 6, 7,
                                      8, 9, 10, 11, 12, 13, 14, 15);
    acc = __builtin_amdgcn_wmma_f32_16x16x32_bf16(false, a, false, b,
                                                  (short)0, acc, false, false);
  }

  int rowbase = tm * 16 + ((lane >= 16) ? 8 : 0);
  int col = tn * 16 + ml;
#pragma unroll
  for (int r = 0; r < 8; r++) {
    int row = rowbase + r;
    if (row < M && col < Nvalid) {
      float v = acc[r] + (bias ? bias[col] : 0.0f);
      if (relu) v = fmaxf(v, 0.0f);
      C[(long long)row * ldc + col] = v;
    }
  }
}

// ------------------------------- pooling -----------------------------------

__global__ void k_scores(const float* __restrict__ x, int ldx,
                         const float* __restrict__ w,
                         float* __restrict__ s, long long rows) {
  long long r = (long long)blockIdx.x * blockDim.x + threadIdx.x;
  if (r >= rows) return;
  float dot = 0.0f, nrm = 0.0f;
  for (int f = 0; f < DF; f++) { float wf = w[f]; dot += x[r * ldx + f] * wf; nrm += wf * wf; }
  s[r] = tanhf(dot * rsqrtf(nrm));
}

// one block (512 threads) per graph: bitonic sort desc-by-score, asc-index ties
__global__ void k_topk(const float* __restrict__ scores, int n, int k,
                       int* __restrict__ mapping, int* __restrict__ idxk,
                       float* __restrict__ svalk) {
  __shared__ float s[512];
  __shared__ int   id[512];
  int g = blockIdx.x, t = threadIdx.x;
  if (t < n) { s[t] = scores[(long long)g * n + t]; mapping[(long long)g * n + t] = -1; }
  else       { s[t] = -__builtin_inff(); }
  id[t] = t;
  __syncthreads();
  for (int kk = 2; kk <= 512; kk <<= 1)
    for (int j = kk >> 1; j > 0; j >>= 1) {
      int i = t ^ j;
      if (i > t) {
        float st = s[t], si = s[i]; int dt = id[t], di = id[i];
        bool tfirst = (st > si) || (st == si && dt < di);
        bool desc = ((t & kk) == 0);
        if (desc != tfirst) { s[t] = si; s[i] = st; id[t] = di; id[i] = dt; }
      }
      __syncthreads();
    }
  if (t < k) {
    int old = id[t];
    mapping[(long long)g * n + old] = g * k + t;
    idxk[g * k + t] = old;
    svalk[g * k + t] = s[t];
  }
}

__global__ void k_gather_scale(const int* __restrict__ idxk, const float* __restrict__ svalk,
                               const float* __restrict__ x, int n, int k,
                               float* __restrict__ xk) {
  long long tid = (long long)blockIdx.x * blockDim.x + threadIdx.x;
  long long tot = (long long)NB * k * DF;
  if (tid >= tot) return;
  int f = (int)(tid % DF);
  long long r = tid / DF;
  int g = (int)(r / k);
  int old = idxk[r];
  xk[r * DF + f] = x[((long long)g * n + old) * DF + f] * svalk[r];
}

__global__ void k_remap(int* __restrict__ src, int* __restrict__ dst, int* __restrict__ valid,
                        const int* __restrict__ mapping, int nE) {
  int e = blockIdx.x * blockDim.x + threadIdx.x;
  if (e >= nE) return;
  if (valid[e]) {
    int ns = mapping[src[e]], nd = mapping[dst[e]];
    if (ns >= 0 && nd >= 0) { src[e] = ns; dst[e] = nd; }
    else { src[e] = 0; dst[e] = 0; valid[e] = 0; }
  } else { src[e] = 0; dst[e] = 0; }
}

// [gmp || gap] -> out[NB, 256]
__global__ void k_readout(const float* __restrict__ xk, int k, float* __restrict__ out) {
  int tid = blockIdx.x * blockDim.x + threadIdx.x;
  if (tid >= NB * DF) return;
  int g = tid / DF, f = tid - g * DF;
  float mx = -__builtin_inff(), sm = 0.0f;
  for (int j = 0; j < k; j++) {
    float v = xk[((long long)g * k + j) * DF + f];
    mx = fmaxf(mx, v); sm += v;
  }
  out[(long long)g * 256 + f] = mx;
  out[(long long)g * 256 + 128 + f] = sm / (float)k;
}

// ------------------------------- head kernels ------------------------------

__global__ void k_combine1(const float* __restrict__ a, const float* __restrict__ b,
                           float* __restrict__ o, int n) {
  int i = blockIdx.x * blockDim.x + threadIdx.x;
  if (i < n) o[i] = fmaxf(a[i], 0.0f) + fmaxf(b[i], 0.0f);
}

__global__ void k_combine2(const float* __restrict__ p0, const float* __restrict__ p1,
                           const float* __restrict__ p2, const float* __restrict__ wr,
                           float* __restrict__ o, int n) {
  int i = blockIdx.x * blockDim.x + threadIdx.x;
  if (i < n)
    o[i] = wr[0] * fmaxf(p0[i], 0.0f) + wr[1] * fmaxf(p1[i], 0.0f) + wr[2] * fmaxf(p2[i], 0.0f);
}

// training-mode BatchNorm1d over rows (biased var, eps=1e-5); optional relu
// before (stats on relu'd values) or after.
__global__ void k_bn(const float* __restrict__ x, float* __restrict__ y,
                     int ld, int rows, int cols,
                     const float* __restrict__ g, const float* __restrict__ be,
                     int relu_before, int relu_after) {
  int c = blockIdx.x * blockDim.x + threadIdx.x;
  if (c >= cols) return;
  float sum = 0.0f;
  for (int r = 0; r < rows; r++) {
    float v = x[(long long)r * ld + c];
    if (relu_before) v = fmaxf(v, 0.0f);
    sum += v;
  }
  float m = sum / (float)rows;
  float vs = 0.0f;
  for (int r = 0; r < rows; r++) {
    float v = x[(long long)r * ld + c];
    if (relu_before) v = fmaxf(v, 0.0f);
    float d = v - m; vs += d * d;
  }
  float inv = rsqrtf(vs / (float)rows + 1e-5f) * g[c];
  for (int r = 0; r < rows; r++) {
    float v = x[(long long)r * ld + c];
    if (relu_before) v = fmaxf(v, 0.0f);
    float o = (v - m) * inv + be[c];
    if (relu_after) o = fmaxf(o, 0.0f);
    y[(long long)r * ld + c] = o;
  }
}

// ------------------------------- driver ------------------------------------

extern "C" void kernel_launch(void* const* d_in, const int* in_sizes, int n_in,
                              void* d_out, int out_size, void* d_ws, size_t ws_size,
                              hipStream_t stream) {
  (void)in_sizes; (void)n_in; (void)out_size; (void)ws_size;

  // ---- input pointers (dict insertion order; params flattened sorted-key) --
  const float* X  = (const float*)d_in[0];   // [65536, 93]
  const int*   EI = (const int*)d_in[1];     // [2, E]
  const float* Aa = (const float*)d_in[3];   // [65536, 43]
  const int*   EG = (const int*)d_in[4];     // [2, E]
  // params: conv1..conv6 {W,b} = 6..17
  const float* convW[6], * convB[6];
  for (int i = 0; i < 6; i++) { convW[i] = (const float*)d_in[6 + 2 * i]; convB[i] = (const float*)d_in[7 + 2 * i]; }
  // fc_final: be,g,l1W,l1b,l2W,l2b = 18..23 ; fc_final1 = 24..29
  // fc_g = 30..35 ; fc_g1 = 36..41 ; pool_w = 42 ; wr = 43
  const float* fcf_be  = (const float*)d_in[18], * fcf_g  = (const float*)d_in[19];
  const float* fcf_l1W = (const float*)d_in[20], * fcf_l1b = (const float*)d_in[21];
  const float* fcf_l2W = (const float*)d_in[22], * fcf_l2b = (const float*)d_in[23];
  const float* fcf1_be = (const float*)d_in[24], * fcf1_g = (const float*)d_in[25];
  const float* fcf1_l1W = (const float*)d_in[26], * fcf1_l1b = (const float*)d_in[27];
  const float* fcf1_l2W = (const float*)d_in[28], * fcf1_l2b = (const float*)d_in[29];
  const float* fcg_be  = (const float*)d_in[30], * fcg_g  = (const float*)d_in[31];
  const float* fcg_l1W = (const float*)d_in[32], * fcg_l1b = (const float*)d_in[33];
  const float* fcg_l2W = (const float*)d_in[34], * fcg_l2b = (const float*)d_in[35];
  const float* fcg1_be = (const float*)d_in[36], * fcg1_g = (const float*)d_in[37];
  const float* fcg1_l1W = (const float*)d_in[38], * fcg1_l1b = (const float*)d_in[39];
  const float* fcg1_l2W = (const float*)d_in[40], * fcg1_l2b = (const float*)d_in[41];
  const float* poolw = (const float*)d_in[42]; // [6,128]
  const float* wr    = (const float*)d_in[43]; // [3]

  float* dout = (float*)d_out;
  float* out_z      = dout;                        // [128,1]
  float* out_xg_all = dout + 128;                  // [128,512]
  float* out_xg1    = dout + 128 + NB * 512;       // [128,512]
  float* out_z1     = dout + 128 + 2 * NB * 512;   // [128,1]

  // ---- workspace bump allocator ------------------------------------------
  char* ws = (char*)d_ws;
  size_t off = 0;
  auto alloc = [&](size_t bytes) -> void* {
    void* p = ws + off;
    off = (off + bytes + 255) & ~(size_t)255;
    return p;
  };
  float* regA   = (float*)alloc((size_t)65536 * 128 * 4);          // AGG / conv out
  float* regD   = (float*)alloc((size_t)NB * KP1 * 128 * 4);       // pooled ping
  float* regE   = (float*)alloc((size_t)NB * KP1 * 128 * 4);       // pooled pong
  __bf16* hbB   = (__bf16*)alloc((size_t)65536 * 128 * 2);         // bf16 GEMM input
  int* srcB     = (int*)alloc((size_t)NEDGE * 4);
  int* dstB     = (int*)alloc((size_t)NEDGE * 4);
  int* validB   = (int*)alloc((size_t)NEDGE * 4);
  float* scoresB = (float*)alloc((size_t)65536 * 4);
  int*   idxkB   = (int*)alloc((size_t)NB * KP1 * 4);
  float* svalkB  = (float*)alloc((size_t)NB * KP1 * 4);
  int*   mapB    = (int*)alloc((size_t)65536 * 4);
  float* outs1   = (float*)alloc((size_t)3 * NB * 256 * 4);
  float* outs2   = (float*)alloc((size_t)3 * NB * 256 * 4);
  float* xgin    = (float*)alloc((size_t)NB * 256 * 4);
  float* xin1    = (float*)alloc((size_t)NB * 256 * 4);
  float* ht1     = (float*)alloc((size_t)NB * 1024 * 4);
  float* ht2     = (float*)alloc((size_t)NB * 1024 * 4);
  __bf16* hbh    = (__bf16*)alloc((size_t)NB * 1024 * 2);

  // bf16 weights
  auto allocw = [&](size_t elems) -> __bf16* { return (__bf16*)alloc(elems * 2); };
  int convKpad[6] = { 96, 128, 128, 64, 128, 128 };
  int convKin[6]  = { 93, 128, 128, 43, 128, 128 };
  __bf16* wbConv[6];
  for (int i = 0; i < 6; i++) wbConv[i] = allocw((size_t)128 * convKpad[i]);
  __bf16* wb_fcg_l1  = allocw((size_t)1024 * 256);
  __bf16* wb_fcg_l2  = allocw((size_t)512 * 1024);
  __bf16* wb_fcf_l1  = allocw((size_t)256 * 512);
  __bf16* wb_fcf_l2  = allocw((size_t)16 * 256);
  __bf16* wb_fcg1_l1 = allocw((size_t)1024 * 256);
  __bf16* wb_fcg1_l2 = allocw((size_t)512 * 1024);
  __bf16* wb_fcf1_l1 = allocw((size_t)256 * 512);
  __bf16* wb_fcf1_l2 = allocw((size_t)16 * 256);

  auto cw = [&](const float* w, __bf16* wb, int nout, int kin, int kpad, int npad) {
    k_conv_w<<<nblk((long long)npad * kpad), LB, 0, stream>>>(w, wb, nout, kin, kpad, npad);
  };
  for (int i = 0; i < 6; i++) cw(convW[i], wbConv[i], 128, convKin[i], convKpad[i], 128);
  cw(fcg_l1W,  wb_fcg_l1,  1024, 256, 256, 1024);
  cw(fcg_l2W,  wb_fcg_l2,  512, 1024, 1024, 512);
  cw(fcf_l1W,  wb_fcf_l1,  256, 512, 512, 256);
  cw(fcf_l2W,  wb_fcf_l2,  1, 256, 256, 16);
  cw(fcg1_l1W, wb_fcg1_l1, 1024, 256, 256, 1024);
  cw(fcg1_l2W, wb_fcg1_l2, 512, 1024, 1024, 512);
  cw(fcf1_l1W, wb_fcf1_l1, 256, 512, 512, 256);
  cw(fcf1_l2W, wb_fcf1_l2, 1, 256, 256, 16);

  auto gemm = [&](const __bf16* Ap, int lda, const __bf16* Bp, int ldb, const float* bias,
                  float* Cp, int ldc, int M, int Npad, int Nvalid, int Kpad, int relu) {
    long long tiles = (long long)((M + 15) / 16) * (Npad / 16);
    dim3 blk(32, 8);
    dim3 grd((unsigned)((tiles + 7) / 8));
    k_gemm<<<grd, blk, 0, stream>>>(Ap, lda, Bp, ldb, bias, Cp, ldc, M, Npad, Nvalid, Kpad, relu);
  };

  // ---- GIN + TopK branch --------------------------------------------------
  auto run_branch = [&](const float* x0, int feat0, int fpad0, const int* ei,
                        const __bf16* const* wconv, const float* const* bconv,
                        const float* pw, float* outs) {
    k_copy_edges<<<nblk(NEDGE), LB, 0, stream>>>(ei, ei + NEDGE, srcB, dstB, validB, NEDGE);
    const float* xcur = x0;
    int ldx = feat0, featc = feat0, kpadc = fpad0, n = NN0;
    float* xkbuf[2] = { regD, regE };
    int ks[3] = { KP1, KP2, KP3 };
    for (int l = 0; l < 3; l++) {
      long long rows = (long long)NB * n;
      k_zero_f32<<<nblk(rows * kpadc), LB, 0, stream>>>(regA, rows * kpadc);
      k_scatter_add<<<nblk((long long)NEDGE * featc), LB, 0, stream>>>(
          xcur, ldx, srcB, dstB, validB, regA, kpadc, NEDGE, featc);
      k_build_hb<<<nblk(rows * kpadc), LB, 0, stream>>>(
          xcur, ldx, regA, kpadc, hbB, kpadc, rows, featc);
      gemm(hbB, kpadc, wconv[l], kpadc, bconv[l], regA, DF, (int)rows, DF, DF, kpadc, 1);
      k_scores<<<nblk(rows), LB, 0, stream>>>(regA, DF, pw + l * DF, scoresB, rows);
      k_topk<<<NB, 512, 0, stream>>>(scoresB, n, ks[l], mapB, idxkB, svalkB);
      float* xk = xkbuf[l & 1];
      k_gather_scale<<<nblk((long long)NB * ks[l] * DF), LB, 0, stream>>>(
          idxkB, svalkB, regA, n, ks[l], xk);
      k_remap<<<nblk(NEDGE), LB, 0, stream>>>(srcB, dstB, validB, mapB, NEDGE);
      k_readout<<<nblk(NB * DF), LB, 0, stream>>>(xk, ks[l], outs + (long long)l * NB * 256);
      xcur = xk; ldx = DF; featc = DF; kpadc = DF; n = ks[l];
    }
  };

  const __bf16* wc1[3] = { wbConv[0], wbConv[1], wbConv[2] };
  const float*  bc1[3] = { convB[0], convB[1], convB[2] };
  const __bf16* wc2[3] = { wbConv[3], wbConv[4], wbConv[5] };
  const float*  bc2[3] = { convB[3], convB[4], convB[5] };
  run_branch(X,  93, 96, EI, wc1, bc1, poolw,            outs1);
  run_branch(Aa, 43, 64, EG, wc2, bc2, poolw + 3 * DF,   outs2);

  // ---- head 1: xg_all = fc_g(relu(o0)+relu(o1)); z = fc_final(xg_all) -----
  int nh = NB * 256;
  k_combine1<<<nblk(nh), LB, 0, stream>>>(outs1, outs1 + NB * 256, xgin, nh);
  k_build_hb<<<nblk((long long)NB * 256), LB, 0, stream>>>(xgin, 256, nullptr, 0, hbh, 256, NB, 256);
  gemm(hbh, 256, wb_fcg_l1, 256, fcg_l1b, ht1, 1024, NB, 1024, 1024, 256, 0);
  k_bn<<<nblk(1024), LB, 0, stream>>>(ht1, ht1, 1024, NB, 1024, fcg_g, fcg_be, 0, 1);
  k_build_hb<<<nblk((long long)NB * 1024), LB, 0, stream>>>(ht1, 1024, nullptr, 0, hbh, 1024, NB, 1024);
  gemm(hbh, 1024, wb_fcg_l2, 1024, fcg_l2b, out_xg_all, 512, NB, 512, 512, 1024, 0);
  k_build_hb<<<nblk((long long)NB * 512), LB, 0, stream>>>(out_xg_all, 512, nullptr, 0, hbh, 512, NB, 512);
  gemm(hbh, 512, wb_fcf_l1, 512, fcf_l1b, ht2, 256, NB, 256, 256, 512, 0);
  k_bn<<<nblk(256), LB, 0, stream>>>(ht2, ht2, 256, NB, 256, fcf_g, fcf_be, 0, 1);
  k_build_hb<<<nblk((long long)NB * 256), LB, 0, stream>>>(ht2, 256, nullptr, 0, hbh, 256, NB, 256);
  gemm(hbh, 256, wb_fcf_l2, 256, fcf_l2b, out_z, 1, NB, 16, 1, 256, 0);

  // ---- head 2: xg1 = fc_g1(w·relu(p)); z1 = fc_final1(xg1) ---------------
  k_combine2<<<nblk(nh), LB, 0, stream>>>(outs2, outs2 + NB * 256, outs2 + 2 * NB * 256, wr, xin1, nh);
  k_build_hb<<<nblk((long long)NB * 256), LB, 0, stream>>>(xin1, 256, nullptr, 0, hbh, 256, NB, 256);
  gemm(hbh, 256, wb_fcg1_l1, 256, fcg1_l1b, ht1, 1024, NB, 1024, 1024, 256, 1); // relu then BN
  k_bn<<<nblk(1024), LB, 0, stream>>>(ht1, ht1, 1024, NB, 1024, fcg1_g, fcg1_be, 0, 0);
  k_build_hb<<<nblk((long long)NB * 1024), LB, 0, stream>>>(ht1, 1024, nullptr, 0, hbh, 1024, NB, 1024);
  gemm(hbh, 1024, wb_fcg1_l2, 1024, fcg1_l2b, out_xg1, 512, NB, 512, 512, 1024, 0);
  k_build_hb<<<nblk((long long)NB * 512), LB, 0, stream>>>(out_xg1, 512, nullptr, 0, hbh, 512, NB, 512);
  gemm(hbh, 512, wb_fcf1_l1, 512, fcf1_l1b, ht2, 256, NB, 256, 256, 512, 0);
  k_bn<<<nblk(256), LB, 0, stream>>>(ht2, ht2, 256, NB, 256, fcf1_g, fcf1_be, 0, 1);
  k_build_hb<<<nblk((long long)NB * 256), LB, 0, stream>>>(ht2, 256, nullptr, 0, hbh, 256, NB, 256);
  gemm(hbh, 256, wb_fcf1_l2, 256, fcf1_l2b, out_z1, 1, NB, 16, 1, 256, 0);
}
